// GCNEncoder_12438225289578
// MI455X (gfx1250) — matrix operand order repaired and checked by reading, not compiled, not run on hardware
//
#include <hip/hip_runtime.h>
#include <hip/hip_bf16.h>

typedef __attribute__((ext_vector_type(16))) __bf16 v16bf;
typedef __attribute__((ext_vector_type(8)))  float  v8f;
typedef __attribute__((ext_vector_type(8)))  __bf16 v8bf;

constexpr int kN = 8192;   // nodes
constexpr int kF = 128;    // F_IN
constexpr int kH = 64;     // HID
constexpr int kO = 64;     // OUT
constexpr int kNO = kN * kO;           // 524288 elements per output tensor
constexpr float kSlope = 0.01f;        // leaky-relu

constexpr int kKT   = 128;             // K-tile staged in LDS per iteration
constexpr int kBPad = 144;             // LDS row stride (bf16): 144*2B=288B=72 DW, 72%64=8

// ---------------------------------------------------------------------------
// B-spline (GRID_SIZE=5, SPLINE_ORDER=3): grid g[i] = 0.4*i - 2.2, i=0..11.
// ---------------------------------------------------------------------------
__device__ __forceinline__ void bspline8(float x, float* __restrict__ bs) {
    float b[11];
#pragma unroll
    for (int j = 0; j < 11; ++j) {
        float gj  = 0.4f * j - 2.2f;
        float gj1 = gj + 0.4f;
        b[j] = (x >= gj && x < gj1) ? 1.0f : 0.0f;
    }
#pragma unroll
    for (int k = 1; k <= 3; ++k) {
        float inv = 1.0f / (0.4f * (float)k);
#pragma unroll
        for (int j = 0; j < 10; ++j) {
            if (j + k < 11) {
                float gj   = 0.4f * j - 2.2f;
                float gjk  = 0.4f * (j + k) - 2.2f;
                b[j] = (x - gj) * inv * b[j] + (gjk + 0.4f - x) * inv * b[j + 1];
            }
        }
    }
#pragma unroll
    for (int j = 0; j < 8; ++j) bs[j] = b[j];
}

__device__ __forceinline__ float silu_f(float x) {
    return x / (1.0f + __expf(-x));
}

// ---------------------------------------------------------------------------
// Kernel 1: s1t[h][n] = bf16((x@W1)[n][h]); k1[n][h] = KAN layer1.
// ---------------------------------------------------------------------------
__global__ __launch_bounds__(256) void pre1_kernel(
    const float* __restrict__ x,     // [kN][kF]
    const float* __restrict__ W1,    // [kF][kH]
    const float* __restrict__ bw1,   // [kH][kF]
    const float* __restrict__ sw1,   // [kH][kF][8]
    const float* __restrict__ sc1,   // [kH][kF]
    __bf16* __restrict__ s1t,        // [kH][kN]
    float* __restrict__ k1out)       // [kN][kH]
{
    const int row = blockIdx.x * 256 + threadIdx.x;
    const float* xr = x + (size_t)row * kF;

#pragma unroll 1
    for (int hc = 0; hc < 4; ++hc) {
        float aA[16], aK[16];
#pragma unroll
        for (int i = 0; i < 16; ++i) { aA[i] = 0.f; aK[i] = 0.f; }

#pragma unroll 1
        for (int f = 0; f < kF; ++f) {
            float xv = xr[f];
            float sv = silu_f(xv);
            float bs[8];
            bspline8(xv, bs);
#pragma unroll
            for (int hh = 0; hh < 16; ++hh) {
                int h = hc * 16 + hh;
                aA[hh] = fmaf(xv, W1[f * kH + h], aA[hh]);
                const float* sw = sw1 + ((size_t)(h * kF + f)) * 8;
                float sp = 0.f;
#pragma unroll
                for (int k = 0; k < 8; ++k) sp = fmaf(bs[k], sw[k], sp);
                aK[hh] = fmaf(sv, bw1[h * kF + f],
                          fmaf(sp, sc1[h * kF + f], aK[hh]));
            }
        }
#pragma unroll
        for (int hh = 0; hh < 16; ++hh) {
            int h = hc * 16 + hh;
            s1t[(size_t)h * kN + row] = (__bf16)aA[hh];
            k1out[(size_t)row * kH + h] = aK[hh];
        }
    }
}

// ---------------------------------------------------------------------------
// Kernel 3: s2t[o][n] = bf16((g1@W2)[n][o]); k2[n][o] = KAN layer2 on k1.
// ---------------------------------------------------------------------------
__global__ __launch_bounds__(256) void pre2_kernel(
    const float* __restrict__ g1,    // [kN][kH]
    const float* __restrict__ k1,    // [kN][kH]
    const float* __restrict__ W2,    // [kH][kO]
    const float* __restrict__ bw2,   // [kO][kH]
    const float* __restrict__ sw2,   // [kO][kH][8]
    const float* __restrict__ sc2,   // [kO][kH]
    __bf16* __restrict__ s2t,        // [kO][kN]
    float* __restrict__ k2out)       // [kN][kO]
{
    const int row = blockIdx.x * 256 + threadIdx.x;
    const float* gr = g1 + (size_t)row * kH;
    const float* kr = k1 + (size_t)row * kH;

#pragma unroll 1
    for (int hc = 0; hc < 4; ++hc) {
        float aA[16], aK[16];
#pragma unroll
        for (int i = 0; i < 16; ++i) { aA[i] = 0.f; aK[i] = 0.f; }

#pragma unroll 1
        for (int f = 0; f < kH; ++f) {
            float gv = gr[f];
            float kv = kr[f];
            float sv = silu_f(kv);
            float bs[8];
            bspline8(kv, bs);
#pragma unroll
            for (int hh = 0; hh < 16; ++hh) {
                int o = hc * 16 + hh;
                aA[hh] = fmaf(gv, W2[f * kO + o], aA[hh]);
                const float* sw = sw2 + ((size_t)(o * kH + f)) * 8;
                float sp = 0.f;
#pragma unroll
                for (int k = 0; k < 8; ++k) sp = fmaf(bs[k], sw[k], sp);
                aK[hh] = fmaf(sv, bw2[o * kH + f],
                          fmaf(sp, sc2[o * kH + f], aK[hh]));
            }
        }
#pragma unroll
        for (int hh = 0; hh < 16; ++hh) {
            int o = hc * 16 + hh;
            s2t[(size_t)o * kN + row] = (__bf16)aA[hh];
            k2out[(size_t)row * kO + o] = aK[hh];
        }
    }
}

// ---------------------------------------------------------------------------
// Adjacency GEMM: out[n][c] = leaky( sum_k adj[n][k]*Bt[c][k] + bias[c] )
//  - A: adj fp32 streamed from HBM, cvt->bf16 in regs (compiler pipelines the
//    4x b128 loads a chunk ahead with partial loadcnt waits).
//  - B: staged into LDS with CDNA5 async copies (global_load_async_to_lds_b128,
//    ASYNCcnt), double-buffered per 128-wide K-tile; fragments read via ds_load
//    so WMMAs never stall on L2 round trips.
//  - v_wmma_f32_16x16x32_bf16, fp32 accumulation over K=8192.
// 4 waves/WG, wave owns 16 rows x 64 cols. 128 WGs.
// ---------------------------------------------------------------------------
__global__ __launch_bounds__(128) void adj_gemm_kernel(
    const float*  __restrict__ adj,   // [kN][kN]
    const __bf16* __restrict__ Bt,    // [64][kN] transposed bf16 support
    const float*  __restrict__ bias,  // [64]
    float* __restrict__ out)          // [kN][64]
{
    __shared__ __bf16 sB[2][64][kBPad];   // 2 x 64 cols x 128(+pad) K, ~36 KB

    const int tid   = threadIdx.x;
    const int lane  = tid & 31;
    const int wave  = tid >> 5;
    const int mrow  = lane & 15;
    const int khalf = lane >> 4;
    const int r0    = (blockIdx.x * 4 + wave) * 16;

    const float* arow = adj + (size_t)(r0 + mrow) * kN;

    // staging assignment: thread -> (column, 16B segment)
    const int scol0 = tid >> 4;     // 0..7 (column within i-group of 8)
    const int sseg  = tid & 15;     // 0..15 (16B segment within column)

    // ---- stage tile 0 into buffer 0 (async copies, ASYNCcnt) ----
#pragma unroll
    for (int i = 0; i < 8; ++i) {
        const int col = i * 8 + scol0;
        unsigned lds  = (unsigned)(size_t)&sB[0][col][sseg * 8];
        unsigned long long ga =
            (unsigned long long)(size_t)(Bt + (size_t)col * kN + sseg * 8);
        asm volatile("global_load_async_to_lds_b128 %0, %1, off"
                     :: "v"(lds), "v"(ga) : "memory");
    }
    asm volatile("s_wait_asynccnt 0x0" ::: "memory");
    __syncthreads();

    v8f acc0 = {}, acc1 = {}, acc2 = {}, acc3 = {};

#pragma unroll 1
    for (int kt = 0; kt < kN; kt += kKT) {
        const int buf = (kt >> 7) & 1;

        // ---- issue async staging of NEXT tile into the other buffer ----
        if (kt + kKT < kN) {
#pragma unroll
            for (int i = 0; i < 8; ++i) {
                const int col = i * 8 + scol0;
                unsigned lds  = (unsigned)(size_t)&sB[buf ^ 1][col][sseg * 8];
                unsigned long long ga = (unsigned long long)(size_t)
                    (Bt + (size_t)col * kN + (kt + kKT) + sseg * 8);
                asm volatile("global_load_async_to_lds_b128 %0, %1, off"
                             :: "v"(lds), "v"(ga) : "memory");
            }
        }

        // ---- compute current tile: 4 chunks of K=32 ----
#pragma unroll
        for (int kk = 0; kk < 4; ++kk) {
            const int kb = kt + kk * 32;

            // stream-ahead prefetch of adj into GL2 (speculative, OOB-safe)
            __builtin_prefetch(arow + kb + 2048, 0, 0);

            // A fragment (16x32 bf16): lane<16 -> K {0..7,16..23},
            //                          lane>=16 -> K {8..15,24..31}
            const float4 a01 = *(const float4*)(arow + kb + khalf * 8);
            const float4 a23 = *(const float4*)(arow + kb + khalf * 8 + 4);
            const float4 a45 = *(const float4*)(arow + kb + 16 + khalf * 8);
            const float4 a67 = *(const float4*)(arow + kb + 16 + khalf * 8 + 4);
            v16bf A;
            A[0]  = (__bf16)a01.x; A[1]  = (__bf16)a01.y;
            A[2]  = (__bf16)a01.z; A[3]  = (__bf16)a01.w;
            A[4]  = (__bf16)a23.x; A[5]  = (__bf16)a23.y;
            A[6]  = (__bf16)a23.z; A[7]  = (__bf16)a23.w;
            A[8]  = (__bf16)a45.x; A[9]  = (__bf16)a45.y;
            A[10] = (__bf16)a45.z; A[11] = (__bf16)a45.w;
            A[12] = (__bf16)a67.x; A[13] = (__bf16)a67.y;
            A[14] = (__bf16)a67.z; A[15] = (__bf16)a67.w;

            // B fragments from LDS (two 16B halves per lane: K {8h..8h+7, 16+8h..16+8h+7})
            const int kofs = kk * 32 + khalf * 8;
            v16bf B0 = __builtin_shufflevector(*(const v8bf*)&sB[buf][mrow +  0][kofs], *(const v8bf*)&sB[buf][mrow +  0][kofs + 16], 0,1,2,3,4,5,6,7,8,9,10,11,12,13,14,15);
            v16bf B1 = __builtin_shufflevector(*(const v8bf*)&sB[buf][mrow + 16][kofs], *(const v8bf*)&sB[buf][mrow + 16][kofs + 16], 0,1,2,3,4,5,6,7,8,9,10,11,12,13,14,15);
            v16bf B2 = __builtin_shufflevector(*(const v8bf*)&sB[buf][mrow + 32][kofs], *(const v8bf*)&sB[buf][mrow + 32][kofs + 16], 0,1,2,3,4,5,6,7,8,9,10,11,12,13,14,15);
            v16bf B3 = __builtin_shufflevector(*(const v8bf*)&sB[buf][mrow + 48][kofs], *(const v8bf*)&sB[buf][mrow + 48][kofs + 16], 0,1,2,3,4,5,6,7,8,9,10,11,12,13,14,15);

            acc0 = __builtin_amdgcn_wmma_f32_16x16x32_bf16(
                       false, A, false, B0, (short)0, acc0, false, false);
            acc1 = __builtin_amdgcn_wmma_f32_16x16x32_bf16(
                       false, A, false, B1, (short)0, acc1, false, false);
            acc2 = __builtin_amdgcn_wmma_f32_16x16x32_bf16(
                       false, A, false, B2, (short)0, acc2, false, false);
            acc3 = __builtin_amdgcn_wmma_f32_16x16x32_bf16(
                       false, A, false, B3, (short)0, acc3, false, false);
        }

        // next-tile async copies done + everyone finished reading this buffer
        asm volatile("s_wait_asynccnt 0x0" ::: "memory");
        __syncthreads();
    }

    // C layout: VGPR v, lanes 0-15 -> row r0+v, col=lane; lanes 16-31 -> row r0+8+v
    const int rbase = r0 + khalf * 8;
    const float bv0 = bias[mrow];
    const float bv1 = bias[16 + mrow];
    const float bv2 = bias[32 + mrow];
    const float bv3 = bias[48 + mrow];
#pragma unroll
    for (int v = 0; v < 8; ++v) {
        size_t o = (size_t)(rbase + v) * kO + mrow;
        float t;
        t = acc0[v] + bv0; out[o]      = t > 0.f ? t : kSlope * t;
        t = acc1[v] + bv1; out[o + 16] = t > 0.f ? t : kSlope * t;
        t = acc2[v] + bv2; out[o + 32] = t > 0.f ? t : kSlope * t;
        t = acc3[v] + bv3; out[o + 48] = t > 0.f ? t : kSlope * t;
    }
}

// ---------------------------------------------------------------------------
// Attention pool over the two branches (softmax over 2 logits).
// ---------------------------------------------------------------------------
__global__ __launch_bounds__(256) void pool_kernel(
    const float* __restrict__ g2,    // [kN][kO]
    const float* __restrict__ k2,    // [kN][kO]
    const float* __restrict__ attw,  // [kO]
    float* __restrict__ gk)          // [kN][kO]
{
    const int row = blockIdx.x * 256 + threadIdx.x;
    const float* gr = g2 + (size_t)row * kO;
    const float* kr = k2 + (size_t)row * kO;
    float sg = 0.f, sk = 0.f;
#pragma unroll 8
    for (int c = 0; c < kO; ++c) {
        float w = attw[c];
        sg = fmaf(gr[c], w, sg);
        sk = fmaf(kr[c], w, sk);
    }
    float m  = fmaxf(sg, sk);
    float eg = __expf(sg - m);
    float ek = __expf(sk - m);
    float wg = eg / (eg + ek);
    float wk = 1.0f - wg;
#pragma unroll 8
    for (int c = 0; c < kO; ++c)
        gk[(size_t)row * kO + c] = fmaf(wg, gr[c], wk * kr[c]);
}

// ---------------------------------------------------------------------------
extern "C" void kernel_launch(void* const* d_in, const int* in_sizes, int n_in,
                              void* d_out, int out_size, void* d_ws, size_t ws_size,
                              hipStream_t stream) {
    (void)in_sizes; (void)n_in; (void)out_size; (void)ws_size;
    const float* x    = (const float*)d_in[0];
    const float* adj  = (const float*)d_in[1];
    const float* W1   = (const float*)d_in[2];
    const float* b1   = (const float*)d_in[3];
    const float* W2   = (const float*)d_in[4];
    const float* b2   = (const float*)d_in[5];
    const float* bw1  = (const float*)d_in[6];
    const float* sw1  = (const float*)d_in[7];
    const float* sc1  = (const float*)d_in[8];
    const float* bw2  = (const float*)d_in[9];
    const float* sw2  = (const float*)d_in[10];
    const float* sc2  = (const float*)d_in[11];
    const float* attw = (const float*)d_in[12];

    float* out = (float*)d_out;
    float* g1  = out;               // [kN][64]
    float* g2  = out + 1 * kNO;
    float* k1  = out + 2 * kNO;
    float* k2  = out + 3 * kNO;
    float* gk  = out + 4 * kNO;

    __bf16* s1t = (__bf16*)d_ws;                          // 1 MB
    __bf16* s2t = (__bf16*)((char*)d_ws + (1u << 20));    // 1 MB

    pre1_kernel<<<kN / 256, 256, 0, stream>>>(x, W1, bw1, sw1, sc1, s1t, k1);
    adj_gemm_kernel<<<kN / 64, 128, 0, stream>>>(adj, s1t, b1, g1);
    pre2_kernel<<<kN / 256, 256, 0, stream>>>(g1, k1, W2, bw2, sw2, sc2, s2t, k2);
    adj_gemm_kernel<<<kN / 64, 128, 0, stream>>>(adj, s2t, b2, g2);
    pool_kernel<<<kN / 256, 256, 0, stream>>>(g2, k2, attw, gk);
}
